// MultiHeadAttention_32031866093611
// MI455X (gfx1250) — matrix operand order repaired and checked by reading, not compiled, not run on hardware
//
#include <hip/hip_runtime.h>

// ---------------------------------------------------------------------------
// Types for CDNA5 WMMA (wave32)
// ---------------------------------------------------------------------------
typedef __attribute__((ext_vector_type(16))) __bf16 v16bf;
typedef __attribute__((ext_vector_type(8)))  __bf16 v8bf;
typedef __attribute__((ext_vector_type(4)))  __bf16 v4bf;
typedef __attribute__((ext_vector_type(8)))  float  v8f;
typedef __attribute__((ext_vector_type(4)))  float  v4f;
typedef __attribute__((ext_vector_type(4)))  int    v4i;

#define AS1 __attribute__((address_space(1)))
#define AS3 __attribute__((address_space(3)))

#define BATCH 8
#define SEQ   1024
#define DMODEL 1024
#define NHEAD 16
#define DHEAD 64
#define MROWS (BATCH * SEQ)          // 8192
#define LDP   40                     // k-stride for 32-wide bf16 LDS rows (80B)
#define KLDP  80                     // k-stride for 64-wide bf16 LDS rows (160B)
#define SLDP  36                     // col-stride for 32-wide f32 LDS rows (144B)

#if __has_builtin(__builtin_amdgcn_global_load_async_to_lds_b128) && \
    __has_builtin(__builtin_amdgcn_s_wait_asynccnt)
#define HAVE_ASYNC 1
#else
#define HAVE_ASYNC 0
#endif

// Per-lane 16B copy global -> LDS. Async (ASYNCcnt) when available.
static __device__ inline void async16(const __bf16* g, __bf16* l) {
#if HAVE_ASYNC
  __builtin_amdgcn_global_load_async_to_lds_b128((AS1 v4i*)g, (AS3 v4i*)l, 0, 0);
#else
  *(v8bf*)l = *(const v8bf*)g;
#endif
}
static __device__ inline void wait_async() {
#if HAVE_ASYNC
  __builtin_amdgcn_s_wait_asynccnt(0);
#endif
}

static __device__ inline v8f wmma_bf16(v16bf a, v16bf b, v8f c) {
  return __builtin_amdgcn_wmma_f32_16x16x32_bf16(
      false, a, false, b, (short)0, c, false, false);
}

static __device__ inline v16bf combine8(v8bf lo, v8bf hi) {
  return __builtin_shufflevector(lo, hi, 0, 1, 2, 3, 4, 5, 6, 7,
                                         8, 9, 10, 11, 12, 13, 14, 15);
}

// A-fragment (16x32 bf16), row-major source:
//   elements 0..15 = K 16*half .. 16*half+15 (contiguous)
static __device__ inline v16bf load_a_frag(const __bf16* row, int half) {
  return combine8(*(const v8bf*)(row + 16 * half),
                  *(const v8bf*)(row + 16 * half + 8));
}

// B-fragment (32x16 bf16): lane = column n, inner K = 16*half + e contiguous.
static __device__ inline v16bf load_b_frag(const __bf16* col) {
  return combine8(*(const v8bf*)(col), *(const v8bf*)(col + 8));
}

// ---------------------------------------------------------------------------
// 1a) straight f32 -> bf16 convert (activations)
// ---------------------------------------------------------------------------
__global__ __launch_bounds__(256) void cvt_kernel(const float* __restrict__ s,
                                                  __bf16* __restrict__ d, int n) {
  int i = (blockIdx.x * 256 + threadIdx.x) * 4;
  if (i < n) {
    float4 f = *(const float4*)(s + i);
    v4bf o;
    o[0] = (__bf16)f.x; o[1] = (__bf16)f.y;
    o[2] = (__bf16)f.z; o[3] = (__bf16)f.w;
    *(v4bf*)(d + i) = o;
  }
}

// ---------------------------------------------------------------------------
// 1b) f32 -> bf16 convert + transpose: Wt[n][k] = W[k][n]  (1024x1024)
// ---------------------------------------------------------------------------
__global__ __launch_bounds__(256) void cvt_t_kernel(const float* __restrict__ s,
                                                    __bf16* __restrict__ d) {
  __shared__ float tile[32][33];
  int n0 = blockIdx.x * 32, k0 = blockIdx.y * 32;
  int tx = threadIdx.x & 31, ty = threadIdx.x >> 5;  // ty 0..7
#pragma unroll
  for (int i = ty; i < 32; i += 8)
    tile[i][tx] = s[(size_t)(k0 + i) * DMODEL + n0 + tx];
  __syncthreads();
#pragma unroll
  for (int i = ty; i < 32; i += 8)
    d[(size_t)(n0 + i) * DMODEL + k0 + tx] = (__bf16)tile[tx][i];
}

// ---------------------------------------------------------------------------
// 2) Tiled bf16 GEMM: C[M,N] = A[M,K] * Wt[N,K]^T + bias
//    MODE 0: bf16 out, Q/K head-split layout   [bh][s][dk]
//    MODE 2: bf16 out, V transposed layout     [bh][dk][s]
//    MODE 1: f32 out, row-major                [m][n]
//    128 threads (4 waves); tile 64x64; k-step 32; async double-buffered LDS.
// ---------------------------------------------------------------------------
template <int MODE>
__global__ __launch_bounds__(128) void gemm_bf16_kernel(
    const __bf16* __restrict__ A, const __bf16* __restrict__ Wt,
    const float* __restrict__ bias, __bf16* __restrict__ outh,
    float* __restrict__ outf, int M, int N, int K) {
  __shared__ __bf16 lA[2][64 * LDP];
  __shared__ __bf16 lB[2][64 * LDP];

  const int t = threadIdx.x;
  const int w = t >> 5, lane = t & 31, ln = lane & 15, half = lane >> 4;
  const int m0 = blockIdx.x * 64, n0 = blockIdx.y * 64;
  const int c = t & 3, r = t >> 2;  // c: 16B k-chunk, r: 0..31

  v8f acc[4];
#pragma unroll
  for (int j = 0; j < 4; ++j)
#pragma unroll
    for (int i = 0; i < 8; ++i) acc[j][i] = 0.0f;

  const __bf16* gA = A + (size_t)(m0 + r) * K + 8 * c;
  const __bf16* gB = Wt + (size_t)(n0 + r) * K + 8 * c;

  // stage tile at k0 into buffer buf (4 async 16B copies per thread)
  auto stage = [&](int k0, int buf) {
    async16(gA + k0, &lA[buf][r * LDP + 8 * c]);
    async16(gA + (size_t)32 * K + k0, &lA[buf][(r + 32) * LDP + 8 * c]);
    async16(gB + k0, &lB[buf][r * LDP + 8 * c]);
    async16(gB + (size_t)32 * K + k0, &lB[buf][(r + 32) * LDP + 8 * c]);
  };

  stage(0, 0);
  const int nIter = K / 32;
  for (int it = 0; it < nIter; ++it) {
    wait_async();
    __syncthreads();  // tile `it` staged by everyone; everyone done with it-1
    if (it + 1 < nIter) stage((it + 1) * 32, (it + 1) & 1);

    const __bf16* la = &lA[it & 1][(16 * w + ln) * LDP];
    v16bf a = load_a_frag(la, half);
#pragma unroll
    for (int j = 0; j < 4; ++j) {
      v16bf b = load_b_frag(&lB[it & 1][(16 * j + ln) * LDP + 16 * half]);
      acc[j] = wmma_bf16(a, b, acc[j]);
    }
  }

  // epilogue: C layout row = r + 8*half, col = 16*j + ln
#pragma unroll
  for (int j = 0; j < 4; ++j) {
    int n = n0 + 16 * j + ln;
    float bv = bias[n];
#pragma unroll
    for (int rr = 0; rr < 8; ++rr) {
      int m = m0 + 16 * w + rr + 8 * half;
      float val = acc[j][rr] + bv;
      if (MODE == 1) {
        outf[(size_t)m * N + n] = val;
      } else {
        int b = m >> 10, s = m & (SEQ - 1);
        int h = n >> 6, dk = n & (DHEAD - 1);
        size_t bh = (size_t)b * NHEAD + h;
        if (MODE == 0)
          outh[(bh * SEQ + s) * DHEAD + dk] = (__bf16)val;
        else  // MODE 2: V transposed
          outh[(bh * DHEAD + dk) * SEQ + s] = (__bf16)val;
      }
    }
  }
}

// ---------------------------------------------------------------------------
// 3) Fused attention (flash-style online softmax), one (b,h,64 q-rows)/block.
//    - K/V tiles async-staged, double-buffered
//    - softmax stats computed lane-local after one f32 LDS relayout
//    - row sums via P @ ones WMMA (lands directly in C layout)
//    - P A-fragment = each lane's own contiguous half-row (k = 16*half + e), in registers
// ---------------------------------------------------------------------------
__global__ __launch_bounds__(128) void attn_kernel(
    const __bf16* __restrict__ Q, const __bf16* __restrict__ Km,
    const __bf16* __restrict__ Vt, const int* __restrict__ prefix,
    __bf16* __restrict__ attn) {
  __shared__ __bf16 lK[2][32 * KLDP];   // [k-row][dk], padded
  __shared__ __bf16 lV[2][64 * LDP];    // [dk][k], padded
  __shared__ float  lS[4][16 * SLDP];   // per-wave raw score tile (f32)
  __shared__ float  lSc[4][16];         // per-wave per-row scale factors

  const int t = threadIdx.x;
  const int w = t >> 5, lane = t & 31, ln = lane & 15, half = lane >> 4;
  const int blk = blockIdx.x;
  const int qb = blk & 15;
  const int bh = blk >> 4;
  const int b = bh >> 4, h = bh & 15;
  const int q0 = qb * 64;

  const __bf16* Qb = Q + (size_t)bh * SEQ * DHEAD;
  const __bf16* Kb = Km + (size_t)bh * SEQ * DHEAD;
  const __bf16* Vb = Vt + (size_t)bh * DHEAD * SEQ;  // transposed [dk][s]
  const int pfx = prefix[b];

  // Q A-fragments for this wave's 16 rows, loaded once
  const __bf16* qp = Qb + (size_t)(q0 + 16 * w + ln) * DHEAD;
  v16bf aq0 = load_a_frag(qp, half);
  v16bf aq1 = load_a_frag(qp + 32, half);

  // all-ones B operand for row-sum WMMA
  v16bf ones;
#pragma unroll
  for (int i = 0; i < 16; ++i) ones[i] = (__bf16)1.0f;

  // R-layout row tracked by this lane (both lanes of a pair track same row)
  const int qrow = q0 + 16 * w + ln;
  const bool qf0 = qrow < pfx;
  float rmR = -1.0e30f;   // running row max (R layout)
  float rl[8];            // running row sums (C layout)
  v8f o[4];
#pragma unroll
  for (int rr = 0; rr < 8; ++rr) rl[rr] = 0.0f;
#pragma unroll
  for (int j = 0; j < 4; ++j)
#pragma unroll
    for (int i = 0; i < 8; ++i) o[j][i] = 0.0f;

  float* lSw = lS[w];
  float* lScw = lSc[w];

  // stage K tile (32 x 64) + V tile (64 x 32): 256 chunks each, 2 per thread
  auto stageKV = [&](int kt, int buf) {
#pragma unroll
    for (int pass = 0; pass < 2; ++pass) {
      int ck = t + pass * 128;
      int krow = ck >> 3, kwi = ck & 7;
      async16(Kb + (size_t)(kt * 32 + krow) * DHEAD + kwi * 8,
              &lK[buf][krow * KLDP + kwi * 8]);
      int vrow = ck >> 2, vwi = ck & 3;
      async16(Vb + (size_t)vrow * SEQ + kt * 32 + vwi * 8,
              &lV[buf][vrow * LDP + vwi * 8]);
    }
  };

  stageKV(0, 0);
  for (int kt = 0; kt < SEQ / 32; ++kt) {
    wait_async();
    __syncthreads();
    if (kt + 1 < SEQ / 32) stageKV(kt + 1, (kt + 1) & 1);
    const __bf16* lKb = lK[kt & 1];
    const __bf16* lVb = lV[kt & 1];

    v8f z;
#pragma unroll
    for (int i = 0; i < 8; ++i) z[i] = 0.0f;

    // scores: 16x32 tile = two 16x16 accumulators, dk reduced in 2 steps
    const __bf16* kp0 = &lKb[ln * KLDP + 16 * half];
    const __bf16* kp1 = &lKb[(16 + ln) * KLDP + 16 * half];
    v8f s0 = wmma_bf16(aq0, load_b_frag(kp0), z);
    s0 = wmma_bf16(aq1, load_b_frag(kp0 + 32), s0);
    v8f s1 = wmma_bf16(aq0, load_b_frag(kp1), z);
    s1 = wmma_bf16(aq1, load_b_frag(kp1 + 32), s1);

    // C-layout -> R-layout relayout of raw f32 scores through LDS
#pragma unroll
    for (int rr = 0; rr < 8; ++rr) {
      lSw[(rr + 8 * half) * SLDP + ln] = s0[rr];
      lSw[(rr + 8 * half) * SLDP + 16 + ln] = s1[rr];
    }
    v4f sv[4];
#pragma unroll
    for (int i = 0; i < 4; ++i)
      sv[i] = *(const v4f*)&lSw[ln * SLDP + 16 * half + 4 * i];

    // lane-local mask + row max (each lane owns a contiguous half-row)
    const int kbase = kt * 32 + 16 * half;
    float tm = -1.0e30f;
#pragma unroll
    for (int e = 0; e < 16; ++e) {
      float v = sv[e >> 2][e & 3];
      if (!qf0 && (kbase + e) < qrow) v = -1.0e30f;
      sv[e >> 2][e & 3] = v;
      tm = fmaxf(tm, v);
    }
    float nm = fmaxf(rmR, tm);
    float sc = __expf(rmR - nm);
    rmR = nm;

    // lane-local exp + bf16 convert
    v16bf own;
#pragma unroll
    for (int e = 0; e < 16; ++e) {
      float v = sv[e >> 2][e & 3];
      float ev = (v < -1.0e29f) ? 0.0f : __expf(v - nm);
      own[e] = (__bf16)ev;
    }

    // P A-fragment = own half-row as is (k = 16*half + e); xor-16 exchange result unused
    v8bf lo8 = __builtin_shufflevector(own, own, 0, 1, 2, 3, 4, 5, 6, 7);
    v8bf hi8 = __builtin_shufflevector(own, own, 8, 9, 10, 11, 12, 13, 14, 15);
    v8bf send = half ? lo8 : hi8;
    union U { v8bf v; int i[4]; };
    U su, ru;
    su.v = send;
#pragma unroll
    for (int i2 = 0; i2 < 4; ++i2) ru.i[i2] = __shfl_xor(su.i[i2], 16, 32);
    v8bf recv = ru.v;
    v16bf pa = combine8(lo8, hi8);

    // distribute per-row scale factors R-layout -> C-layout
    if (half == 0) lScw[ln] = sc;
    float scC[8];
#pragma unroll
    for (int rr = 0; rr < 8; ++rr) scC[rr] = lScw[rr + 8 * half];

    // row sums of P via WMMA against all-ones (result already in C layout)
    v8f ts8 = wmma_bf16(pa, ones, z);

#pragma unroll
    for (int rr = 0; rr < 8; ++rr) {
      rl[rr] = rl[rr] * scC[rr] + ts8[rr];
#pragma unroll
      for (int j = 0; j < 4; ++j) o[j][rr] = o[j][rr] * scC[rr];
    }

    // o += P (16x32) * V (32x64)
#pragma unroll
    for (int j = 0; j < 4; ++j) {
      v16bf bv = load_b_frag(&lVb[(16 * j + ln) * LDP + 16 * half]);
      o[j] = wmma_bf16(pa, bv, o[j]);
    }
  }

  // epilogue: normalize, write with the reference's head-interleave scramble
#pragma unroll
  for (int rr = 0; rr < 8; ++rr) {
    float inv = 1.0f / rl[rr];
    int q = q0 + 16 * w + rr + 8 * half;
    int sp = h * 64 + (q >> 4);
    int dbase = (q & 15) * 64;
#pragma unroll
    for (int j = 0; j < 4; ++j) {
      float val = o[j][rr] * inv;
      attn[((size_t)b * SEQ + sp) * DMODEL + dbase + 16 * j + ln] = (__bf16)val;
    }
  }
}

// ---------------------------------------------------------------------------
// Host launcher
// ---------------------------------------------------------------------------
extern "C" void kernel_launch(void* const* d_in, const int* in_sizes, int n_in,
                              void* d_out, int out_size, void* d_ws,
                              size_t ws_size, hipStream_t stream) {
  (void)in_sizes; (void)n_in; (void)out_size; (void)ws_size;

  const float* x    = (const float*)d_in[0];
  const int* prefix = (const int*)d_in[1];
  const float* Wq = (const float*)d_in[2];
  const float* bq = (const float*)d_in[3];
  const float* Wk = (const float*)d_in[4];
  const float* bk = (const float*)d_in[5];
  const float* Wv = (const float*)d_in[6];
  const float* bv = (const float*)d_in[7];
  const float* Wo = (const float*)d_in[8];
  const float* bo = (const float*)d_in[9];
  float* out = (float*)d_out;

  const size_t MD = (size_t)MROWS * DMODEL;   // 8388608
  const size_t DD = (size_t)DMODEL * DMODEL;  // 1048576

  __bf16* ws = (__bf16*)d_ws;
  __bf16* xb    = ws;          // activations bf16
  __bf16* wqt   = xb + MD;     // transposed bf16 weights [n][k]
  __bf16* wkt   = wqt + DD;
  __bf16* wvt   = wkt + DD;
  __bf16* wot   = wvt + DD;
  __bf16* Qb    = wot + DD;    // [bh][s][dk]
  __bf16* Kbuf  = Qb + MD;     // [bh][s][dk]
  __bf16* Vbuf  = Kbuf + MD;   // [bh][dk][s] (transposed)
  __bf16* attnb = Vbuf + MD;   // scrambled attention output, bf16

  // 1) convert x; convert+transpose weights
  cvt_kernel<<<(unsigned)((MD / 4 + 255) / 256), 256, 0, stream>>>(x, xb, (int)MD);
  dim3 tg(DMODEL / 32, DMODEL / 32);
  cvt_t_kernel<<<tg, 256, 0, stream>>>(Wq, wqt);
  cvt_t_kernel<<<tg, 256, 0, stream>>>(Wk, wkt);
  cvt_t_kernel<<<tg, 256, 0, stream>>>(Wv, wvt);
  cvt_t_kernel<<<tg, 256, 0, stream>>>(Wo, wot);

  // 2) Q/K/V projections (V emitted transposed)
  dim3 gg(MROWS / 64, DMODEL / 64);
  gemm_bf16_kernel<0><<<gg, 128, 0, stream>>>(xb, wqt, bq, Qb, nullptr,
                                              MROWS, DMODEL, DMODEL);
  gemm_bf16_kernel<0><<<gg, 128, 0, stream>>>(xb, wkt, bk, Kbuf, nullptr,
                                              MROWS, DMODEL, DMODEL);
  gemm_bf16_kernel<2><<<gg, 128, 0, stream>>>(xb, wvt, bv, Vbuf, nullptr,
                                              MROWS, DMODEL, DMODEL);

  // 3) fused masked attention
  attn_kernel<<<BATCH * NHEAD * (SEQ / 64), 128, 0, stream>>>(Qb, Kbuf, Vbuf,
                                                              prefix, attnb);

  // 4) output projection -> f32
  gemm_bf16_kernel<1><<<gg, 128, 0, stream>>>(attnb, wot, bo, nullptr, out,
                                              MROWS, DMODEL, DMODEL);
}
